// TopicClassificationGNN_47845935677472
// MI455X (gfx1250) — matrix operand order repaired and checked by reading, not compiled, not run on hardware
//
#include <hip/hip_runtime.h>
#include <hip/hip_bf16.h>
#include <math.h>

typedef __attribute__((ext_vector_type(16))) _Float16 v16h;
typedef __attribute__((ext_vector_type(8)))  _Float16 v8h;
typedef __attribute__((ext_vector_type(8)))  float    v8f;

#define D_C 64          // per-head channel count (D_H in reference)
#define N_CLS 50

// ---------------------------------------------------------------------------
// fp32 -> f16 elementwise convert (activation rows / weight staging)
// ---------------------------------------------------------------------------
__global__ void cvt_f16(const float* __restrict__ A, _Float16* __restrict__ Ah, size_t n) {
    size_t i = (size_t)blockIdx.x * blockDim.x + threadIdx.x;
    if (i < n) Ah[i] = (_Float16)A[i];
}

// ---------------------------------------------------------------------------
// Pack B[K,Nc] (fp32 row-major) into WMMA B-fragment order, f16:
//   Bp[((kb*(Nc/16) + tn)*32 + lane)*16 + j] = B[(kb*32 + (lane>>4)*16 + j)*Nc
//                                                + tn*16 + (lane&15)]
// so each lane's fragment is 16 contiguous f16 (32 bytes).
// ---------------------------------------------------------------------------
__global__ void pack_B_f16(const float* __restrict__ B, _Float16* __restrict__ Bp,
                           int K, int Nc) {
    int gid = blockIdx.x * blockDim.x + threadIdx.x;
    if (gid >= K * Nc) return;
    const int ntiles = Nc >> 4;
    const int j    = gid & 15;
    const int lane = (gid >> 4) & 31;
    const int rest = gid >> 9;
    const int tn   = rest % ntiles;
    const int kb   = rest / ntiles;
    const int col  = tn * 16 + (lane & 15);
    const int k    = kb * 32 + (lane >> 4) * 16 + j;
    Bp[gid] = (_Float16)B[(size_t)k * Nc + col];
}

// ---------------------------------------------------------------------------
// GEMM  C[M,Nc] = A[M,K] @ B[K,Nc], A in f16 row-major, B pre-packed f16.
// One wave computes a 16x64 output strip (4 accumulators, shared A fragment).
// M%16==0, K%32==0, Nc%64==0. Tile index is wave-uniform -> EXEC all-ones.
// ---------------------------------------------------------------------------
__global__ void gemm_wmma_f16p(const _Float16* __restrict__ Ah,
                               const _Float16* __restrict__ Bp,
                               float* __restrict__ C,
                               int M, int K, int Nc) {
    const int lane = threadIdx.x & 31;
    const int wave = threadIdx.x >> 5;
    const int nstrips = Nc >> 6;           // 64-wide strips
    const int ntiles  = Nc >> 4;
    const int strip = blockIdx.x * (blockDim.x >> 5) + wave;
    if (strip >= (M >> 4) * nstrips) return;     // uniform per wave

    const int tm  = strip / nstrips;
    const int tn4 = strip - tm * nstrips;        // strip col / 64
    const int l15 = lane & 15;
    const int kh  = lane >> 4;

    const int arow = tm * 16 + l15;
    const _Float16* Abase = Ah + (size_t)arow * K + kh * 16;
    const _Float16* Bbase = Bp + ((size_t)(tn4 * 4) * 32 + lane) * 16;
    const size_t BkbStride = (size_t)ntiles * 32 * 16;   // per k-block
    const size_t BtnStride = (size_t)32 * 16;            // per 16-col tile

    v8f acc0 = {}, acc1 = {}, acc2 = {}, acc3 = {};
    for (int kb = 0; kb < (K >> 5); ++kb) {
        // A fragment: two contiguous 8xf16 chunks (K 0..7 and 8..15, +kh*16)
        const _Float16* Ap = Abase + kb * 32;
        v8h a0 = *(const v8h*)(Ap);
        v8h a1 = *(const v8h*)(Ap + 8);
        v16h a = __builtin_shufflevector(a0, a1, 0, 1, 2, 3, 4, 5, 6, 7,
                                                 8, 9, 10, 11, 12, 13, 14, 15);
        // B fragments: one contiguous 32B load per lane per sub-tile
        const _Float16* Bk = Bbase + kb * BkbStride;
        v16h b0 = *(const v16h*)(Bk);
        v16h b1 = *(const v16h*)(Bk + BtnStride);
        v16h b2 = *(const v16h*)(Bk + 2 * BtnStride);
        v16h b3 = *(const v16h*)(Bk + 3 * BtnStride);
        acc0 = __builtin_amdgcn_wmma_f32_16x16x32_f16(false, a, false, b0, (short)0, acc0, false, false);
        acc1 = __builtin_amdgcn_wmma_f32_16x16x32_f16(false, a, false, b1, (short)0, acc1, false, false);
        acc2 = __builtin_amdgcn_wmma_f32_16x16x32_f16(false, a, false, b2, (short)0, acc2, false, false);
        acc3 = __builtin_amdgcn_wmma_f32_16x16x32_f16(false, a, false, b3, (short)0, acc3, false, false);
    }

    // C layout per tile: VGPR i -> row tm*16 + kh*8 + i, col = tilecol + l15
    const int crow0 = tm * 16 + kh * 8;
    float* Cb = C + (size_t)crow0 * Nc + tn4 * 64 + l15;
    #pragma unroll
    for (int i = 0; i < 8; ++i) {
        float* Cr = Cb + (size_t)i * Nc;
        Cr[0]  = acc0[i];
        Cr[16] = acc1[i];
        Cr[32] = acc2[i];
        Cr[48] = acc3[i];
    }
}

// ---------------------------------------------------------------------------
// a_s[n,h] = sum_c h[n,h,c]*att_s[h,c] ; a_d likewise
// ---------------------------------------------------------------------------
__global__ void att_scores(const float* __restrict__ h,
                           const float* __restrict__ att_s,
                           const float* __restrict__ att_d,
                           float* __restrict__ a_s, float* __restrict__ a_d,
                           int N, int H) {
    int gid = blockIdx.x * blockDim.x + threadIdx.x;
    if (gid >= N * H) return;
    const int hh = gid % H;
    const float* hp = h + (size_t)gid * D_C;
    const float* ws = att_s + hh * D_C;
    const float* wd = att_d + hh * D_C;
    float ss = 0.f, sd = 0.f;
    #pragma unroll 8
    for (int c = 0; c < D_C; ++c) { ss += hp[c] * ws[c]; sd += hp[c] * wd[c]; }
    a_s[gid] = ss; a_d[gid] = sd;
}

__global__ void seg_init(float* __restrict__ emax, float* __restrict__ denom, int n) {
    int i = blockIdx.x * blockDim.x + threadIdx.x;
    if (i < n) { emax[i] = -1e30f; denom[i] = 0.f; }
}

__global__ void zero_f32(float* __restrict__ p, size_t n) {
    size_t i = (size_t)blockIdx.x * blockDim.x + threadIdx.x;
    if (i < n) p[i] = 0.f;
}

__device__ __forceinline__ void edge_ends(int eid, const int* __restrict__ ei,
                                          int E, int& s, int& d) {
    if (eid < E) { s = ei[eid]; d = ei[E + eid]; }
    else         { s = eid - E; d = eid - E; }      // self loops
}

__device__ __forceinline__ float lrelu02(float v) {
    return v > 0.f ? v : 0.2f * v;
}

// IEEE-ordered float atomic max via integer atomics
__device__ __forceinline__ void atomicMaxF(float* addr, float val) {
    if (val >= 0.f) atomicMax((int*)addr, __float_as_int(val));
    else            atomicMin((unsigned int*)addr, __float_as_uint(val));
}

__global__ void edge_max(const int* __restrict__ ei, int E, int N, int H,
                         const float* __restrict__ a_s, const float* __restrict__ a_d,
                         float* __restrict__ emax) {
    int gid = blockIdx.x * blockDim.x + threadIdx.x;
    const int tot = (E + N) * H;
    if (gid >= tot) return;
    const int eid = gid / H, hh = gid - eid * H;
    int s, d; edge_ends(eid, ei, E, s, d);
    const float e = lrelu02(a_s[s * H + hh] + a_d[d * H + hh]);
    atomicMaxF(&emax[d * H + hh], e);
}

__global__ void edge_sum(const int* __restrict__ ei, int E, int N, int H,
                         const float* __restrict__ a_s, const float* __restrict__ a_d,
                         const float* __restrict__ emax, float* __restrict__ denom) {
    int gid = blockIdx.x * blockDim.x + threadIdx.x;
    const int tot = (E + N) * H;
    if (gid >= tot) return;
    const int eid = gid / H, hh = gid - eid * H;
    int s, d; edge_ends(eid, ei, E, s, d);
    const float e = lrelu02(a_s[s * H + hh] + a_d[d * H + hh]);
    const float p = __expf(e - emax[d * H + hh]);
    atomicAdd(&denom[d * H + hh], p);
}

// One WAVE per (edge, head): 64-wide feature row scattered with
// 2 coalesced float atomics per lane.
__global__ void edge_accum(const int* __restrict__ ei, int E, int N, int H,
                           const float* __restrict__ a_s, const float* __restrict__ a_d,
                           const float* __restrict__ emax, const float* __restrict__ denom,
                           const float* __restrict__ hfeat, float* __restrict__ accum) {
    const int lane = threadIdx.x & 31;
    const int wid = (blockIdx.x * blockDim.x + threadIdx.x) >> 5;
    const int tot = (E + N) * H;
    if (wid >= tot) return;                 // uniform per wave
    const int eid = wid / H, hh = wid - eid * H;
    int s, d; edge_ends(eid, ei, E, s, d);
    const float e = lrelu02(a_s[s * H + hh] + a_d[d * H + hh]);
    const float p = __expf(e - emax[d * H + hh]);
    const float alpha = p / (denom[d * H + hh] + 1e-16f);
    const float* hp = hfeat + ((size_t)s * H + hh) * D_C;
    float* ap = accum + ((size_t)d * H + hh) * D_C;
    atomicAdd(&ap[lane],      hp[lane]      * alpha);
    atomicAdd(&ap[lane + 32], hp[lane + 32] * alpha);
}

// out[n,c] = mean_h accum[n,h,c] + bias[c], optional BN(eval) + ReLU
__global__ void head_reduce(const float* __restrict__ accum,
                            const float* __restrict__ bias,
                            const float* __restrict__ bn_g, const float* __restrict__ bn_b,
                            const float* __restrict__ bn_m, const float* __restrict__ bn_v,
                            float* __restrict__ out, int N, int H, int use_bn) {
    int gid = blockIdx.x * blockDim.x + threadIdx.x;
    if (gid >= N * D_C) return;
    const int n = gid / D_C, c = gid - n * D_C;
    float s = 0.f;
    for (int hh = 0; hh < H; ++hh) s += accum[((size_t)n * H + hh) * D_C + c];
    s = s / (float)H + bias[c];
    if (use_bn) {
        s = (s - bn_m[c]) * bn_g[c] * rsqrtf(bn_v[c] + 1e-5f) + bn_b[c];
        s = fmaxf(s, 0.f);
    }
    out[gid] = s;
}

__global__ void bias_relu(float* __restrict__ z, const float* __restrict__ b, int N) {
    int gid = blockIdx.x * blockDim.x + threadIdx.x;
    if (gid >= N * D_C) return;
    z[gid] = fmaxf(z[gid] + b[gid % D_C], 0.f);
}

// logits = z @ cW2 + cb2 ; softmax per node
__global__ void logits_softmax(const float* __restrict__ z,
                               const float* __restrict__ cW2, const float* __restrict__ cb2,
                               float* __restrict__ out, int N) {
    int n = blockIdx.x * blockDim.x + threadIdx.x;
    if (n >= N) return;
    const float* zn = z + (size_t)n * D_C;
    float zl[D_C];
    #pragma unroll 8
    for (int i = 0; i < D_C; ++i) zl[i] = zn[i];
    float lg[N_CLS];
    float mx = -1e30f;
    for (int c = 0; c < N_CLS; ++c) {
        float s = cb2[c];
        for (int i = 0; i < D_C; ++i) s += zl[i] * cW2[i * N_CLS + c];
        lg[c] = s; mx = fmaxf(mx, s);
    }
    float den = 0.f;
    for (int c = 0; c < N_CLS; ++c) { lg[c] = __expf(lg[c] - mx); den += lg[c]; }
    const float inv = 1.f / den;
    float* op = out + (size_t)n * N_CLS;
    for (int c = 0; c < N_CLS; ++c) op[c] = lg[c] * inv;
}

// ---------------------------------------------------------------------------
static inline int cdiv(int a, int b) { return (a + b - 1) / b; }

static void run_gemm(hipStream_t stream, const float* A, const float* B, float* C,
                     int M, int K, int Nc, _Float16* Ah, _Float16* Bp) {
    size_t na = (size_t)M * K;
    cvt_f16<<<(int)((na + 255) / 256), 256, 0, stream>>>(A, Ah, na);
    pack_B_f16<<<cdiv(K * Nc, 256), 256, 0, stream>>>(B, Bp, K, Nc);
    int strips = (M / 16) * (Nc / 64);
    gemm_wmma_f16p<<<cdiv(strips, 8), 256, 0, stream>>>(Ah, Bp, C, M, K, Nc);
}

static void run_gat(hipStream_t stream, const float* xin, int Kin,
                    const float* W, const float* as_, const float* ad_, const float* bias,
                    int H, const float* bg, const float* bb, const float* bm, const float* bv,
                    int use_bn, const int* ei, int E, int N,
                    float* hfeat, float* accum, float* a_s, float* a_d,
                    float* emax, float* denom, float* xout,
                    _Float16* Ah, _Float16* Bp) {
    const int Nc = H * D_C;
    run_gemm(stream, xin, W, hfeat, N, Kin, Nc, Ah, Bp);   // h = xin @ W
    att_scores<<<cdiv(N * H, 256), 256, 0, stream>>>(hfeat, as_, ad_, a_s, a_d, N, H);
    seg_init<<<cdiv(N * H, 256), 256, 0, stream>>>(emax, denom, N * H);
    {
        size_t na = (size_t)N * H * D_C;
        zero_f32<<<(int)((na + 255) / 256), 256, 0, stream>>>(accum, na);
    }
    const int tot = (E + N) * H;
    edge_max<<<cdiv(tot, 256), 256, 0, stream>>>(ei, E, N, H, a_s, a_d, emax);
    edge_sum<<<cdiv(tot, 256), 256, 0, stream>>>(ei, E, N, H, a_s, a_d, emax, denom);
    edge_accum<<<cdiv(tot * 32, 256), 256, 0, stream>>>(ei, E, N, H, a_s, a_d, emax,
                                                        denom, hfeat, accum);
    head_reduce<<<cdiv(N * D_C, 256), 256, 0, stream>>>(accum, bias, bg, bb, bm, bv,
                                                        xout, N, H, use_bn);
}

extern "C" void kernel_launch(void* const* d_in, const int* in_sizes, int n_in,
                              void* d_out, int out_size, void* d_ws, size_t ws_size,
                              hipStream_t stream) {
    const float* x    = (const float*)d_in[0];
    const int*   ei   = (const int*)d_in[1];
    const float* W1   = (const float*)d_in[2];
    const float* as1  = (const float*)d_in[3];
    const float* ad1  = (const float*)d_in[4];
    const float* b1   = (const float*)d_in[5];
    const float* W2   = (const float*)d_in[6];
    const float* as2  = (const float*)d_in[7];
    const float* ad2  = (const float*)d_in[8];
    const float* b2   = (const float*)d_in[9];
    const float* W3   = (const float*)d_in[10];
    const float* as3  = (const float*)d_in[11];
    const float* ad3  = (const float*)d_in[12];
    const float* b3   = (const float*)d_in[13];
    const float* bn1g = (const float*)d_in[14];
    const float* bn1b = (const float*)d_in[15];
    const float* bn1m = (const float*)d_in[16];
    const float* bn1v = (const float*)d_in[17];
    const float* bn2g = (const float*)d_in[18];
    const float* bn2b = (const float*)d_in[19];
    const float* bn2m = (const float*)d_in[20];
    const float* bn2v = (const float*)d_in[21];
    const float* cW1  = (const float*)d_in[22];
    const float* cb1  = (const float*)d_in[23];
    const float* cW2  = (const float*)d_in[24];
    const float* cb2  = (const float*)d_in[25];

    const int N = in_sizes[0] / 256;     // 50000
    const int E = in_sizes[1] / 2;       // 400000
    float* out = (float*)d_out;

    // workspace layout (float units; all chunks are multiples of 64 floats
    // so every pointer stays 256-byte aligned)
    float* ws = (float*)d_ws;
    size_t off = 0;
    float* hfeat = ws + off; off += (size_t)N * 8 * D_C;   // [N, H*64], max H=8
    float* accum = ws + off; off += (size_t)N * 8 * D_C;
    float* a_s   = ws + off; off += (size_t)N * 8;
    float* a_d   = ws + off; off += (size_t)N * 8;
    float* emax  = ws + off; off += (size_t)N * 8;
    float* denom = ws + off; off += (size_t)N * 8;
    float* xb1   = ws + off; off += (size_t)N * D_C;
    float* xb2   = ws + off; off += (size_t)N * D_C;
    _Float16* Ah = (_Float16*)(ws + off); off += (size_t)N * 256 / 2;  // f16 [N, Kmax]
    _Float16* Bp = (_Float16*)(ws + off); off += (size_t)256 * 512 / 2; // f16 packed W

    // Layer 1: heads=8, BN1+ReLU
    run_gat(stream, x, 256, W1, as1, ad1, b1, 8, bn1g, bn1b, bn1m, bn1v, 1,
            ei, E, N, hfeat, accum, a_s, a_d, emax, denom, xb1, Ah, Bp);
    // Layer 2: heads=4, BN2+ReLU
    run_gat(stream, xb1, D_C, W2, as2, ad2, b2, 4, bn2g, bn2b, bn2m, bn2v, 1,
            ei, E, N, hfeat, accum, a_s, a_d, emax, denom, xb2, Ah, Bp);
    // Layer 3: heads=1, concat (no BN/ReLU)
    run_gat(stream, xb2, D_C, W3, as3, ad3, b3, 1, bn1g, bn1b, bn1m, bn1v, 0,
            ei, E, N, hfeat, accum, a_s, a_d, emax, denom, xb1, Ah, Bp);

    // classifier: z = relu(h @ cW1 + cb1) via WMMA, then logits+softmax
    float* z = hfeat;   // reuse
    run_gemm(stream, xb1, cW1, z, N, D_C, D_C, Ah, Bp);
    bias_relu<<<cdiv(N * D_C, 256), 256, 0, stream>>>(z, cb1, N);
    logits_softmax<<<cdiv(N, 128), 128, 0, stream>>>(z, cW2, cb2, out, N);
}
